// VariationalGAT_33638183862457
// MI455X (gfx1250) — matrix operand order repaired and checked by reading, not compiled, not run on hardware
//
#include <hip/hip_runtime.h>
#include <stdint.h>

typedef __attribute__((ext_vector_type(16))) __bf16        v16bf;
typedef __attribute__((ext_vector_type(8)))  float         v8f;
typedef __attribute__((ext_vector_type(4)))  unsigned int  u32x4;
typedef __attribute__((ext_vector_type(2)))  float         f32x2;
typedef unsigned short u16;

#define NN    8192
#define NFEAT 512
#define NHID  256
#define NLAT  64
#define NCLS  16
#define ALPHA 0.2f

union Frag16 {           // 16 bf16 elems = 8 VGPRs/lane, matches WMMA A/B operand
  v16bf v;
  u32x4 q[2];
  unsigned int u[8];
};

__device__ __forceinline__ u16 f2bf(float f) {
  unsigned int u = __float_as_uint(f);
  u = (u + 0x7FFFu + ((u >> 16) & 1u)) >> 16;   // RNE truncate to bf16
  return (u16)u;
}
__device__ __forceinline__ float bf2f(u16 h) {
  return __uint_as_float(((unsigned int)h) << 16);
}
__device__ __forceinline__ unsigned int pack2(float a, float b) {
  return (unsigned int)f2bf(a) | ((unsigned int)f2bf(b) << 16);
}

// ---------------------------------------------------------------------------
// Pack adj (0/1 int32) into a 1-bit mask: msk[word] bit l = adj[word*32+l]>0
// ---------------------------------------------------------------------------
__global__ void k_pack_adj(const int* __restrict__ adj, unsigned int* __restrict__ msk) {
  int tid  = blockIdx.x * blockDim.x + threadIdx.x;
  int wid  = tid >> 5;
  int lane = tid & 31;
  int a = adj[(size_t)wid * 32 + lane];
  unsigned long long b = __ballot(a > 0);      // wave32: low 32 bits valid
  if (lane == 0) msk[wid] = (unsigned int)b;
}

// ---------------------------------------------------------------------------
// Transpose f32 weight [K][Nc] -> bf16 [Nc][K] (so B K-pairs are contiguous)
// ---------------------------------------------------------------------------
__global__ void k_transpose_bf16(const float* __restrict__ W, u16* __restrict__ Wt,
                                 int K, int Nc) {
  int idx = blockIdx.x * blockDim.x + threadIdx.x;
  if (idx < K * Nc) {
    int k = idx / Nc, n = idx - k * Nc;
    Wt[(size_t)n * K + k] = f2bf(W[idx]);
  }
}

// ---------------------------------------------------------------------------
// Fragment loaders
// A fragment (16x32 bf16): lane<16 row r K{0..7,16..23}; lane>=16 K{8..15,24..31}
// ---------------------------------------------------------------------------
template <bool AF32>
__device__ __forceinline__ void load_afrag(Frag16& af, const void* Aptr,
                                           size_t arowK, int k0, int hf) {
  if (AF32) {
    const float* A = (const float*)Aptr + arowK + k0;
#pragma unroll
    for (int v = 0; v < 8; ++v) {
      int kk = ((v >= 4) ? 16 : 0) + hf * 8 + 2 * (v & 3);
      f32x2 f = *(const f32x2*)(A + kk);
      af.u[v] = pack2(f.x, f.y);
    }
  } else {
    const u16* A = (const u16*)Aptr + arowK + k0 + hf * 8;
    af.q[0] = *(const u32x4*)A;
    af.q[1] = *(const u32x4*)(A + 16);
  }
}

// B fragment (32x16 bf16) from a [Nc][K]-transposed bf16 matrix: 2x16B loads
__device__ __forceinline__ void load_bfrag(Frag16& bf_, const u16* __restrict__ Bt,
                                           size_t rowOff, int k0, int hf) {
  const u16* B = Bt + rowOff + k0 + hf * 8;
  bf_.q[0] = *(const u32x4*)B;
  bf_.q[1] = *(const u32x4*)(B + 16);
}

// ---------------------------------------------------------------------------
// Tiled bf16 WMMA GEMM: C[M][Nn] = A[M][K] @ Bt[Nn][K]^T, output stored as
// bf16 *transposed* Ct[Nn][M]. One wave per 16x64 output tile, software-
// pipelined k-loop (ping-pong fragments, unroll by 64) so loads overlap WMMA.
// grid = (M/16, Nn/64), block = 32.
// ---------------------------------------------------------------------------
template <bool AF32>
__global__ __launch_bounds__(32) void k_gemm_bf16(const void* __restrict__ Aptr,
                                                  const u16* __restrict__ Bt,
                                                  u16* __restrict__ Ct,
                                                  int M, int K) {
  const int lane = threadIdx.x;
  const int r = lane & 15, hf = lane >> 4;
  const int m0 = blockIdx.x * 16;
  const int n0 = blockIdx.y * 64;

  v8f acc[4];
#pragma unroll
  for (int t = 0; t < 4; ++t)
#pragma unroll
    for (int p = 0; p < 8; ++p) acc[t][p] = 0.f;

  const size_t arowK = (size_t)(m0 + r) * K;
  size_t brow[4];
#pragma unroll
  for (int t = 0; t < 4; ++t) brow[t] = (size_t)(n0 + t * 16 + r) * K;

  Frag16 afA, afB, bfA[4], bfB[4];
  load_afrag<AF32>(afA, Aptr, arowK, 0, hf);
#pragma unroll
  for (int t = 0; t < 4; ++t) load_bfrag(bfA[t], Bt, brow[t], 0, hf);

  for (int k0 = 0; k0 < K; k0 += 64) {
    // issue next-chunk loads, then compute current chunk
    load_afrag<AF32>(afB, Aptr, arowK, k0 + 32, hf);
#pragma unroll
    for (int t = 0; t < 4; ++t) load_bfrag(bfB[t], Bt, brow[t], k0 + 32, hf);
#pragma unroll
    for (int t = 0; t < 4; ++t)
      acc[t] = __builtin_amdgcn_wmma_f32_16x16x32_bf16(false, afA.v, false, bfA[t].v,
                                                       (short)0, acc[t], false, false);
    if (k0 + 64 < K) {
      load_afrag<AF32>(afA, Aptr, arowK, k0 + 64, hf);
#pragma unroll
      for (int t = 0; t < 4; ++t) load_bfrag(bfA[t], Bt, brow[t], k0 + 64, hf);
    }
#pragma unroll
    for (int t = 0; t < 4; ++t)
      acc[t] = __builtin_amdgcn_wmma_f32_16x16x32_bf16(false, afB.v, false, bfB[t].v,
                                                       (short)0, acc[t], false, false);
  }

  // D layout: VGPR p, lane<16 -> row p col lane ; lane>=16 -> row p+8 col lane-16
#pragma unroll
  for (int t = 0; t < 4; ++t)
#pragma unroll
    for (int p = 0; p < 8; ++p)
      Ct[(size_t)(n0 + t * 16 + r) * M + (m0 + p + 8 * hf)] = f2bf(acc[t][p]);
}

// ---------------------------------------------------------------------------
// s[i] = Wh[i,:] . a[0:F] ; d[i] = Wh[i,:] . a[F:2F], reading Wht bf16 [F][N]
// ---------------------------------------------------------------------------
__global__ void k_matvec_sd(const u16* __restrict__ Wht, const float* __restrict__ a,
                            float* __restrict__ s, float* __restrict__ d, int F) {
  int i = blockIdx.x * blockDim.x + threadIdx.x;
  if (i >= NN) return;
  float ss = 0.f, dd = 0.f;
  for (int f = 0; f < F; ++f) {
    float w = bf2f(Wht[(size_t)f * NN + i]);   // coalesced across threads
    ss += w * a[f];
    dd += w * a[F + f];
  }
  s[i] = ss;
  d[i] = dd;
}

// ---------------------------------------------------------------------------
// Fused GAT attention for a 16-row block:
//   att = softmax_j( mask ? leakyrelu(s_i + d_j) : -inf );  out = att @ Wh
// Block = 4 waves. Pass1: masked row-max. Pass2: exp weights -> LDS bf16 tile
// (A-fragment layout) -> WMMA against Wh tiles. B loads issued before the
// barrier so global latency overlaps the LDS handoff. MODE 0: ELU + bf16 out;
// MODE 1: f32 out.
// ---------------------------------------------------------------------------
template <int FTILES, int MODE>
__global__ __launch_bounds__(128) void k_gat_attn(const u16* __restrict__ Wht,       // [F][N] bf16
                                                  const float* __restrict__ sRow,
                                                  const float* __restrict__ dCol,
                                                  const unsigned int* __restrict__ msk, // [N][N/32]
                                                  void* __restrict__ outPtr) {
  constexpr int FT4 = FTILES / 4;
  constexpr int F = FTILES * 16;
  const int w = threadIdx.x >> 5;
  const int lane = threadIdx.x & 31;
  const int r = lane & 15, hf = lane >> 4;
  const int i0 = blockIdx.x * 16;
  const int MW = NN / 32;

  __shared__ float sh_s[16];
  __shared__ float sh_part[4][16];
  __shared__ float sh_rm[16];
  __shared__ float sh_rs[16];
  __shared__ u16 wt[16 * 32];   // 16x32 bf16 attention-weight tile

  if (threadIdx.x < 16) sh_s[threadIdx.x] = sRow[i0 + threadIdx.x];
  __syncthreads();

  // ---------------- pass 1: masked row max (each wave sweeps N/4 cols) -----
  float m[16];
#pragma unroll
  for (int i = 0; i < 16; ++i) m[i] = -3.0e38f;
  const int jspan = NN / 4;
  for (int j0 = w * jspan; j0 < (w + 1) * jspan; j0 += 32) {
    float dj = dCol[j0 + lane];
    const unsigned int* mp = msk + (size_t)i0 * MW + (j0 >> 5);
#pragma unroll
    for (int i = 0; i < 16; ++i) {
      unsigned int mw = mp[(size_t)i * MW];
      float e = sh_s[i] + dj;
      e = e > 0.f ? e : ALPHA * e;
      if ((mw >> lane) & 1u) m[i] = fmaxf(m[i], e);
    }
  }
#pragma unroll
  for (int i = 0; i < 16; ++i)
    for (int off = 16; off > 0; off >>= 1)
      m[i] = fmaxf(m[i], __shfl_xor(m[i], off, 32));
  if (lane == 0) {
#pragma unroll
    for (int i = 0; i < 16; ++i) sh_part[w][i] = m[i];
  }
  __syncthreads();
  if (threadIdx.x < 16) {
    int t = threadIdx.x;
    sh_rm[t] = fmaxf(fmaxf(sh_part[0][t], sh_part[1][t]),
                     fmaxf(sh_part[2][t], sh_part[3][t]));
  }
  __syncthreads();

  // ---------------- pass 2: exp weights + WMMA accumulate ------------------
  const int ir0 = w * 4;   // this wave owns rows ir0..ir0+3 of the weight tile
  float ps[4] = {0.f, 0.f, 0.f, 0.f};
  v8f acc[FT4];
#pragma unroll
  for (int t = 0; t < FT4; ++t)
#pragma unroll
    for (int p = 0; p < 8; ++p) acc[t][p] = 0.f;

  float rm4[4], s4[4];
#pragma unroll
  for (int ii = 0; ii < 4; ++ii) {
    rm4[ii] = sh_rm[ir0 + ii];
    s4[ii] = sh_s[ir0 + ii];
  }

  size_t brow[FT4];
#pragma unroll
  for (int t = 0; t < FT4; ++t) brow[t] = (size_t)((w * FT4 + t) * 16 + r) * NN;

  for (int j0 = 0; j0 < NN; j0 += 32) {
    float dj = dCol[j0 + lane];
    const unsigned int* mp = msk + (size_t)(i0 + ir0) * MW + (j0 >> 5);
    __builtin_prefetch(mp + 8, 0, 1);          // global_prefetch_b8, stream mask
#pragma unroll
    for (int ii = 0; ii < 4; ++ii) {
      unsigned int mw = mp[(size_t)ii * MW];
      float e = s4[ii] + dj;
      e = e > 0.f ? e : ALPHA * e;
      float wv = ((mw >> lane) & 1u) ? __expf(e - rm4[ii]) : 0.f;
      ps[ii] += wv;
      wt[(ir0 + ii) * 32 + lane] = f2bf(wv);
    }
    // B fragments don't depend on the wt tile: issue them before the barrier
    Frag16 bf_[FT4];
#pragma unroll
    for (int t = 0; t < FT4; ++t) load_bfrag(bf_[t], Wht, brow[t], j0, hf);
    __syncthreads();
    // A fragment from LDS tile (row-major [16][32] bf16)
    Frag16 af;
    const u16* wr = wt + r * 32 + hf * 8;
    af.q[0] = *(const u32x4*)wr;
    af.q[1] = *(const u32x4*)(wr + 16);
#pragma unroll
    for (int t = 0; t < FT4; ++t)
      acc[t] = __builtin_amdgcn_wmma_f32_16x16x32_bf16(false, af.v, false, bf_[t].v,
                                                       (short)0, acc[t], false, false);
    __syncthreads();   // protect wt before next iteration's writes
  }

  // row sums -> 1/sum, scale, store
#pragma unroll
  for (int ii = 0; ii < 4; ++ii)
    for (int off = 16; off > 0; off >>= 1)
      ps[ii] += __shfl_xor(ps[ii], off, 32);
  if (lane == 0) {
#pragma unroll
    for (int ii = 0; ii < 4; ++ii) sh_rs[ir0 + ii] = ps[ii];
  }
  __syncthreads();
  float ri[8];
#pragma unroll
  for (int p = 0; p < 8; ++p) {
    float rs_ = sh_rs[p + 8 * hf];
    ri[p] = rs_ > 0.f ? 1.f / rs_ : 0.f;
  }
#pragma unroll
  for (int t = 0; t < FT4; ++t) {
    int col = (w * FT4 + t) * 16 + r;
#pragma unroll
    for (int p = 0; p < 8; ++p) {
      float v = acc[t][p] * ri[p];
      int row = i0 + p + 8 * hf;
      if (MODE == 0) {
        v = v > 0.f ? v : (__expf(v) - 1.f);                 // ELU
        ((u16*)outPtr)[(size_t)row * F + col] = f2bf(v);
      } else {
        ((float*)outPtr)[(size_t)row * F + col] = v;
      }
    }
  }
}

// ---------------------------------------------------------------------------
// z = mu + eps * exp(0.5*logvar) ; logits = z @ Wc + bc  (64x16, VALU-trivial)
// ---------------------------------------------------------------------------
__global__ void k_z_logits(const float* __restrict__ mu, const float* __restrict__ lv,
                           const float* __restrict__ eps, const float* __restrict__ Wc,
                           const float* __restrict__ bc, float* __restrict__ logits) {
  int i = blockIdx.x * blockDim.x + threadIdx.x;
  if (i >= NN) return;
  float z[NLAT];
#pragma unroll
  for (int f = 0; f < NLAT; ++f)
    z[f] = mu[(size_t)i * NLAT + f] +
           eps[(size_t)i * NLAT + f] * __expf(0.5f * lv[(size_t)i * NLAT + f]);
  for (int c = 0; c < NCLS; ++c) {
    float a = bc[c];
#pragma unroll
    for (int f = 0; f < NLAT; ++f) a += z[f] * Wc[f * NCLS + c];
    logits[(size_t)i * NCLS + c] = a;
  }
}

// ---------------------------------------------------------------------------
extern "C" void kernel_launch(void* const* d_in, const int* in_sizes, int n_in,
                              void* d_out, int out_size, void* d_ws, size_t ws_size,
                              hipStream_t stream) {
  (void)in_sizes; (void)n_in; (void)out_size; (void)ws_size;

  const float* x   = (const float*)d_in[0];
  const int*   adj = (const int*)d_in[1];
  const float* eps = (const float*)d_in[2];
  const float* W1  = (const float*)d_in[3];
  const float* a1  = (const float*)d_in[4];
  const float* Wmu = (const float*)d_in[5];
  const float* amu = (const float*)d_in[6];
  const float* Wlv = (const float*)d_in[7];
  const float* alv = (const float*)d_in[8];
  const float* Wc  = (const float*)d_in[9];
  const float* bc  = (const float*)d_in[10];

  float* logits = (float*)d_out;                      // [N][16]
  float* muOut  = logits + (size_t)NN * NCLS;         // [N][64]
  float* lvOut  = muOut + (size_t)NN * NLAT;          // [N][64]

  char* p = (char*)d_ws;
  auto alloc = [&](size_t bytes) {
    char* q = p;
    p += (bytes + 255) & ~(size_t)255;
    return q;
  };
  unsigned int* msk = (unsigned int*)alloc((size_t)NN * (NN / 32) * 4);  // 8 MB
  u16* W1t   = (u16*)alloc((size_t)NHID * NFEAT * 2);
  u16* Wmut  = (u16*)alloc((size_t)NLAT * NHID * 2);
  u16* Wlvt  = (u16*)alloc((size_t)NLAT * NHID * 2);
  u16* Wh1t  = (u16*)alloc((size_t)NHID * NN * 2);    // [256][8192]
  u16* Whmut = (u16*)alloc((size_t)NLAT * NN * 2);    // [64][8192]
  u16* Whlvt = (u16*)alloc((size_t)NLAT * NN * 2);
  u16* hbf   = (u16*)alloc((size_t)NN * NHID * 2);    // [8192][256] row-major
  float* s1 = (float*)alloc(NN * 4); float* d1 = (float*)alloc(NN * 4);
  float* s2 = (float*)alloc(NN * 4); float* d2 = (float*)alloc(NN * 4);
  float* s3 = (float*)alloc(NN * 4); float* d3 = (float*)alloc(NN * 4);

  // adj -> 1-bit mask (single read of the 256 MB matrix)
  k_pack_adj<<<((size_t)NN * NN / 32) * 32 / 256, 256, 0, stream>>>(adj, msk);

  // weight transposes to bf16
  k_transpose_bf16<<<(NFEAT * NHID + 255) / 256, 256, 0, stream>>>(W1, W1t, NFEAT, NHID);
  k_transpose_bf16<<<(NHID * NLAT + 255) / 256, 256, 0, stream>>>(Wmu, Wmut, NHID, NLAT);
  k_transpose_bf16<<<(NHID * NLAT + 255) / 256, 256, 0, stream>>>(Wlv, Wlvt, NHID, NLAT);

  // Layer 1: Wh1 = x @ W1 (WMMA), s/d, fused attention + ELU -> h (bf16)
  k_gemm_bf16<true><<<dim3(NN / 16, NHID / 64), 32, 0, stream>>>(x, W1t, Wh1t, NN, NFEAT);
  k_matvec_sd<<<NN / 256, 256, 0, stream>>>(Wh1t, a1, s1, d1, NHID);
  k_gat_attn<NHID / 16, 0><<<NN / 16, 128, 0, stream>>>(Wh1t, s1, d1, msk, hbf);

  // Layers 2/3: Wh = h @ W (WMMA), s/d, fused attention -> mu / logvar (f32)
  k_gemm_bf16<false><<<dim3(NN / 16, NLAT / 64), 32, 0, stream>>>(hbf, Wmut, Whmut, NN, NHID);
  k_gemm_bf16<false><<<dim3(NN / 16, NLAT / 64), 32, 0, stream>>>(hbf, Wlvt, Whlvt, NN, NHID);
  k_matvec_sd<<<NN / 256, 256, 0, stream>>>(Whmut, amu, s2, d2, NLAT);
  k_matvec_sd<<<NN / 256, 256, 0, stream>>>(Whlvt, alv, s3, d3, NLAT);
  k_gat_attn<NLAT / 16, 1><<<NN / 16, 128, 0, stream>>>(Whmut, s2, d2, msk, muOut);
  k_gat_attn<NLAT / 16, 1><<<NN / 16, 128, 0, stream>>>(Whlvt, s3, d3, msk, lvOut);

  // reparameterize + classifier head
  k_z_logits<<<NN / 128, 128, 0, stream>>>(muOut, lvOut, eps, Wc, bc, logits);
}
